// Block_31542239822416
// MI455X (gfx1250) — matrix operand order repaired and checked by reading, not compiled, not run on hardware
//
#include <hip/hip_runtime.h>
#include <hip/hip_bf16.h>

// ---- problem constants (from reference) ----
#define BB   4
#define NN   2048
#define CC   512
#define HH   8
#define DH   64
#define HID  2048
#define ROWS (BB * NN)        // 8192
#define EPSL 1e-5f

typedef _Float16 h16 __attribute__((ext_vector_type(16)));
typedef _Float16 h8  __attribute__((ext_vector_type(8)));
typedef float    f8  __attribute__((ext_vector_type(8)));

static __device__ inline h16 cat8(h8 lo, h8 hi) {
    h16 r;
#pragma unroll
    for (int i = 0; i < 8; ++i) { r[i] = lo[i]; r[8 + i] = hi[i]; }
    return r;
}

static __device__ inline f8 wmma_f16(h16 a, h16 b, f8 c) {
    return __builtin_amdgcn_wmma_f32_16x16x32_f16(
        /*neg_a=*/false, a, /*neg_b=*/false, b,
        /*c_mod=*/(short)0, c, /*reuse_a=*/false, /*reuse_b=*/false);
}

// async global -> LDS copy of 16 bytes per lane (CDNA5 ASYNCcnt path)
static __device__ inline void async_b128_to_lds(const void* lds_ptr, const void* gptr) {
    unsigned           loff = (unsigned)(size_t)lds_ptr;          // low 32 bits = LDS offset
    unsigned long long ga   = (unsigned long long)(size_t)gptr;
    asm volatile("global_load_async_to_lds_b128 %0, %1, off"
                 :: "v"(loff), "v"(ga) : "memory");
}
static __device__ inline void wait_async0() {
    asm volatile("s_wait_asynccnt 0x0" ::: "memory");
}

// ---------------- f32 -> f16 convert ----------------
__global__ void cvt_f16_kernel(const float* __restrict__ src,
                               _Float16* __restrict__ dst, int n) {
    int i = blockIdx.x * blockDim.x + threadIdx.x;
    if (i < n) dst[i] = (_Float16)src[i];
}

// ---------------- LayerNorm (f32 in, f16 out), one row per block ----------------
__global__ __launch_bounds__(256, 1)
void ln_kernel(const float* __restrict__ x,
               const float* __restrict__ w,
               const float* __restrict__ b,
               _Float16* __restrict__ out) {
    __shared__ float red[256];
    const int row = blockIdx.x;
    const int tid = threadIdx.x;
    const float* xr = x + (size_t)row * CC;
    float x0 = xr[tid], x1 = xr[tid + 256];

    red[tid] = x0 + x1;
    __syncthreads();
    for (int s = 128; s > 0; s >>= 1) {
        if (tid < s) red[tid] += red[tid + s];
        __syncthreads();
    }
    float mu = red[0] * (1.0f / CC);
    __syncthreads();

    float d0 = x0 - mu, d1 = x1 - mu;
    red[tid] = d0 * d0 + d1 * d1;
    __syncthreads();
    for (int s = 128; s > 0; s >>= 1) {
        if (tid < s) red[tid] += red[tid + s];
        __syncthreads();
    }
    float rs = rsqrtf(red[0] * (1.0f / CC) + EPSL);

    _Float16* orow = out + (size_t)row * CC;
    orow[tid]       = (_Float16)(d0 * rs * w[tid] + b[tid]);
    orow[tid + 256] = (_Float16)(d1 * rs * w[tid + 256] + b[tid + 256]);
}

// ---------------- WMMA GEMM: Y = act(A[M,K] * W[N,K]^T + bias) + res ----------------
// block = 256 threads = 8 waves (2 M-waves x 4 N-waves); block tile 64 x 256;
// wave tile 32 x 64 (2x4 WMMA 16x16 tiles); K step = 32.
template <bool HASBIAS, bool HASRES, bool GELU, bool OUTF16>
__global__ __launch_bounds__(256, 1)
void gemm_kernel(const _Float16* __restrict__ A,
                 const _Float16* __restrict__ W,
                 const float* __restrict__ bias,
                 const float* __restrict__ res,
                 void* __restrict__ outv,
                 int M, int Nout, int K) {
    __shared__ alignas(32) _Float16 ldsA[64 * 32];
    __shared__ alignas(32) _Float16 ldsB[256 * 32];

    const int tid  = threadIdx.x;
    const int lane = tid & 31;
    const int wave = tid >> 5;
    const int mw   = wave >> 2;       // 0..1
    const int nw   = wave & 3;        // 0..3
    const int m0   = blockIdx.x * 64;
    const int n0   = blockIdx.y * 256;
    const int lm   = lane & 15;
    const int lh   = lane >> 4;       // 0 or 1

    f8 acc[2][4];
#pragma unroll
    for (int i = 0; i < 2; ++i)
#pragma unroll
        for (int j = 0; j < 4; ++j) acc[i][j] = (f8)0.0f;

    const int rowA = tid >> 1, chA = tid & 1;   // A staging assignment (tid<128)

    for (int k0 = 0; k0 < K; k0 += 32) {
        __syncthreads();
        // async stage A tile: 64 rows x 32 halves
        if (tid < 128) {
            async_b128_to_lds(&ldsA[rowA * 32 + chA * 16],
                              A + (size_t)(m0 + rowA) * K + k0 + chA * 16);
        }
        // async stage B tile: 256 rows x 32 halves (W rows are K-contiguous)
#pragma unroll
        for (int ii = 0; ii < 2; ++ii) {
            int idx = tid + ii * 256;
            int row = idx >> 1, ch = idx & 1;
            async_b128_to_lds(&ldsB[row * 32 + ch * 16],
                              W + (size_t)(n0 + row) * K + k0 + ch * 16);
        }
        // prefetch next K-step into L2/WGP while this one computes
        if (k0 + 32 < K) {
            if (tid < 128)
                __builtin_prefetch(A + (size_t)(m0 + rowA) * K + k0 + 32 + chA * 16, 0, 0);
            __builtin_prefetch(W + (size_t)(n0 + tid) * K + k0 + 32, 0, 0);
        }
        wait_async0();
        __syncthreads();

        h16 aF[2];
#pragma unroll
        for (int i = 0; i < 2; ++i) {
            int mr = mw * 32 + i * 16 + lm;
            h8 lo = *(const h8*)(&ldsA[mr * 32 + lh * 16]);
            h8 hi = *(const h8*)(&ldsA[mr * 32 + lh * 16 + 8]);
            aF[i] = cat8(lo, hi);
        }
#pragma unroll
        for (int j = 0; j < 4; ++j) {
            int nr = nw * 64 + j * 16 + lm;
            h16 bF = *(const h16*)(&ldsB[nr * 32 + lh * 16]);
#pragma unroll
            for (int i = 0; i < 2; ++i) acc[i][j] = wmma_f16(aF[i], bF, acc[i][j]);
        }
    }

    // epilogue: C layout: element (m,n): vgpr r = m%8, lane = n + 16*(m/8)
#pragma unroll
    for (int i = 0; i < 2; ++i) {
#pragma unroll
        for (int j = 0; j < 4; ++j) {
#pragma unroll
            for (int r = 0; r < 8; ++r) {
                int row = m0 + mw * 32 + i * 16 + r + (lh << 3);
                int col = n0 + nw * 64 + j * 16 + lm;
                float v = acc[i][j][r];
                if (HASBIAS) v += bias[col];
                if (GELU)    v = 0.5f * v * (1.0f + erff(v * 0.70710678118f));
                if (HASRES)  v += res[(size_t)row * Nout + col];
                if (OUTF16)  ((_Float16*)outv)[(size_t)row * Nout + col] = (_Float16)v;
                else         ((float*)outv)[(size_t)row * Nout + col] = v;
            }
        }
    }
}

// ---------------- Flash attention ----------------
// grid: B*H*(N/128) blocks; block = 256 thr = 8 waves; wave owns 16 queries.
// qkv layout: [ROWS, 3C] f16, row = b*N+n, q at [h*64..], k at [512+h*64..], v at [1024+h*64..]
__global__ __launch_bounds__(256, 1)
void attn_kernel(const _Float16* __restrict__ qkv,
                 const float* __restrict__ rpb,
                 _Float16* __restrict__ att) {
    __shared__ alignas(32) _Float16 ldsK[32 * 64];   // [key][d]
    __shared__ alignas(32) _Float16 ldsVT[64 * 32];  // [d][key]
    __shared__ alignas(32) _Float16 ldsP[8 * 16 * 32];

    const int tid  = threadIdx.x;
    const int lane = tid & 31;
    const int wave = tid >> 5;
    const int lm   = lane & 15;
    const int lh   = lane >> 4;

    const int qblk = blockIdx.x & 15;      // 16 blocks of 128 queries per (b,h)
    const int bh   = blockIdx.x >> 4;
    const int b    = bh >> 3;
    const int h    = bh & 7;
    const int q0   = qblk * 128 + wave * 16;

    // Q fragments (K-dim = d; two 16x16x32 chunks)
    const _Float16* qrow = qkv + (size_t)(b * NN + q0 + lm) * (3 * CC) + h * DH;
    h16 aq[2];
#pragma unroll
    for (int kc = 0; kc < 2; ++kc) {
        h8 lo = *(const h8*)(qrow + kc * 32 + lh * 16);
        h8 hi = *(const h8*)(qrow + kc * 32 + lh * 16 + 8);
        aq[kc] = cat8(lo, hi);
    }

    float mI[8], lI[8];
    f8 o[4];
#pragma unroll
    for (int r = 0; r < 8; ++r) { mI[r] = -3.0e38f; lI[r] = 0.0f; }
#pragma unroll
    for (int dt = 0; dt < 4; ++dt) o[dt] = (f8)0.0f;

    const int keyS = tid >> 2 & 31 ? (tid & 127) >> 2 : (tid & 127) >> 2; // (kept simple below)

    for (int kb = 0; kb < NN; kb += 32) {
        __syncthreads();
        // cooperative staging of 32 keys (K async row-major, V transposed via regs)
        if (tid < 128) {
            int key = tid >> 2, ch = tid & 3;
            async_b128_to_lds(&ldsK[key * 64 + ch * 16],
                              qkv + (size_t)(b * NN + kb + key) * (3 * CC)
                                  + CC + h * DH + ch * 16);
            if (kb + 32 < NN)   // prefetch next chunk's K rows
                __builtin_prefetch(qkv + (size_t)(b * NN + kb + 32 + key) * (3 * CC)
                                       + CC + h * DH + ch * 16, 0, 0);
            wait_async0();
        } else {
            int tt = tid - 128;
            int key = tt >> 2, ch = tt & 3;
            h16 vv = *(const h16*)(qkv + (size_t)(b * NN + kb + key) * (3 * CC)
                                   + 2 * CC + h * DH + ch * 16);
#pragma unroll
            for (int i = 0; i < 16; ++i)
                ldsVT[(ch * 16 + i) * 32 + key] = vv[i];
            if (kb + 32 < NN)   // prefetch next chunk's V rows
                __builtin_prefetch(qkv + (size_t)(b * NN + kb + 32 + key) * (3 * CC)
                                       + 2 * CC + h * DH + ch * 16, 0, 0);
        }
        __syncthreads();

        // S = Q K^T for two 16-key tiles
        float sc[2][8];
#pragma unroll
        for (int t = 0; t < 2; ++t) {
            f8 c = (f8)0.0f;
#pragma unroll
            for (int kc = 0; kc < 2; ++kc) {
                h16 bK = *(const h16*)(&ldsK[(t * 16 + lm) * 64 + kc * 32 + lh * 16]);
                c = wmma_f16(aq[kc], bK, c);
            }
            int key = kb + t * 16 + lm;
#pragma unroll
            for (int r = 0; r < 8; ++r) {
                int mrow = r + (lh << 3);
                int idx = key - (q0 + mrow) + (NN - 1);
                sc[t][r] = c[r] * 0.125f + rpb[idx * HH + h];
            }
        }

        // online softmax update (rows live on 16-lane N-groups)
        float rm[8], rs[8], corr[8];
#pragma unroll
        for (int r = 0; r < 8; ++r) rm[r] = fmaxf(sc[0][r], sc[1][r]);
#pragma unroll
        for (int mk = 1; mk < 16; mk <<= 1)
#pragma unroll
            for (int r = 0; r < 8; ++r) rm[r] = fmaxf(rm[r], __shfl_xor(rm[r], mk, 32));
#pragma unroll
        for (int r = 0; r < 8; ++r) {
            float mn = fmaxf(mI[r], rm[r]);
            corr[r] = __expf(mI[r] - mn);
            mI[r] = mn;
            sc[0][r] = __expf(sc[0][r] - mn);
            sc[1][r] = __expf(sc[1][r] - mn);
            rs[r] = sc[0][r] + sc[1][r];
        }
#pragma unroll
        for (int mk = 1; mk < 16; mk <<= 1)
#pragma unroll
            for (int r = 0; r < 8; ++r) rs[r] += __shfl_xor(rs[r], mk, 32);
#pragma unroll
        for (int r = 0; r < 8; ++r) lI[r] = lI[r] * corr[r] + rs[r];
#pragma unroll
        for (int dt = 0; dt < 4; ++dt)
#pragma unroll
            for (int r = 0; r < 8; ++r) o[dt][r] *= corr[r];

        // P: C-layout f32 -> A-layout f16 via per-wave LDS round trip
        _Float16* pP = &ldsP[wave * 512];
#pragma unroll
        for (int t = 0; t < 2; ++t)
#pragma unroll
            for (int r = 0; r < 8; ++r)
                pP[(r + (lh << 3)) * 32 + t * 16 + lm] = (_Float16)sc[t][r];
        h8 plo = *(const h8*)(&pP[lm * 32 + lh * 16]);
        h8 phi = *(const h8*)(&pP[lm * 32 + lh * 16 + 8]);
        h16 aP = cat8(plo, phi);

        // O += P V (4 d-tiles)
#pragma unroll
        for (int dt = 0; dt < 4; ++dt) {
            h16 bV = *(const h16*)(&ldsVT[(dt * 16 + lm) * 32 + lh * 16]);
            o[dt] = wmma_f16(aP, bV, o[dt]);
        }
    }
    (void)keyS;

    // normalize + store f16 [ROWS, C]
#pragma unroll
    for (int dt = 0; dt < 4; ++dt)
#pragma unroll
        for (int r = 0; r < 8; ++r) {
            int row = q0 + r + (lh << 3);
            int col = h * DH + dt * 16 + lm;
            att[(size_t)(b * NN + row) * CC + col] = (_Float16)(o[dt][r] / lI[r]);
        }
}

// ---------------- launch ----------------
extern "C" void kernel_launch(void* const* d_in, const int* in_sizes, int n_in,
                              void* d_out, int out_size, void* d_ws, size_t ws_size,
                              hipStream_t stream) {
    const float* x      = (const float*)d_in[0];
    const float* qkv_w  = (const float*)d_in[1];
    const float* proj_w = (const float*)d_in[2];
    const float* proj_b = (const float*)d_in[3];
    const float* rpb    = (const float*)d_in[4];
    const float* n1_w   = (const float*)d_in[5];
    const float* n1_b   = (const float*)d_in[6];
    const float* n2_w   = (const float*)d_in[7];
    const float* n2_b   = (const float*)d_in[8];
    const float* fc1_w  = (const float*)d_in[9];
    const float* fc1_b  = (const float*)d_in[10];
    const float* fc2_w  = (const float*)d_in[11];
    const float* fc2_b  = (const float*)d_in[12];
    float* out = (float*)d_out;

    char* w = (char*)d_ws;
    _Float16* wQKV = (_Float16*)(w + 0);          // 3C x C
    _Float16* wPRJ = (_Float16*)(w + 1572864);    // C x C
    _Float16* wF1  = (_Float16*)(w + 2097152);    // HID x C
    _Float16* wF2  = (_Float16*)(w + 4194304);    // C x HID
    _Float16* h1   = (_Float16*)(w + 6291456);    // ROWS x C
    _Float16* qkv  = (_Float16*)(w + 14680064);   // ROWS x 3C
    _Float16* att  = (_Float16*)(w + 39845888);   // ROWS x C
    float*    y1   = (float*)   (w + 48234496);   // ROWS x C
    _Float16* h2   = (_Float16*)(w + 65011712);   // ROWS x C
    _Float16* g    = (_Float16*)(w + 73400320);   // ROWS x HID

    // weight conversion
    cvt_f16_kernel<<<(3 * CC * CC + 255) / 256, 256, 0, stream>>>(qkv_w, wQKV, 3 * CC * CC);
    cvt_f16_kernel<<<(CC * CC + 255) / 256, 256, 0, stream>>>(proj_w, wPRJ, CC * CC);
    cvt_f16_kernel<<<(HID * CC + 255) / 256, 256, 0, stream>>>(fc1_w, wF1, HID * CC);
    cvt_f16_kernel<<<(CC * HID + 255) / 256, 256, 0, stream>>>(fc2_w, wF2, CC * HID);

    // LN1 -> h1 (f16)
    ln_kernel<<<ROWS, 256, 0, stream>>>(x, n1_w, n1_b, h1);

    // qkv = h1 @ qkv_w^T  (f16 out)
    gemm_kernel<false, false, false, true>
        <<<dim3(ROWS / 64, (3 * CC) / 256), 256, 0, stream>>>(
            h1, wQKV, nullptr, nullptr, qkv, ROWS, 3 * CC, CC);

    // flash attention -> att (f16)
    attn_kernel<<<BB * HH * (NN / 128), 256, 0, stream>>>(qkv, rpb, att);

    // y1 = x + att @ proj_w^T + proj_b  (f32 out)
    gemm_kernel<true, true, false, false>
        <<<dim3(ROWS / 64, CC / 256), 256, 0, stream>>>(
            att, wPRJ, proj_b, x, y1, ROWS, CC, CC);

    // LN2 -> h2 (f16)
    ln_kernel<<<ROWS, 256, 0, stream>>>(y1, n2_w, n2_b, h2);

    // g = gelu(h2 @ fc1_w^T + fc1_b)  (f16 out)
    gemm_kernel<true, false, true, true>
        <<<dim3(ROWS / 64, HID / 256), 256, 0, stream>>>(
            h2, wF1, fc1_b, nullptr, g, ROWS, HID, CC);

    // out = y1 + g @ fc2_w^T + fc2_b  (f32 out)
    gemm_kernel<true, true, false, false>
        <<<dim3(ROWS / 64, CC / 256), 256, 0, stream>>>(
            g, wF2, fc2_b, y1, out, ROWS, CC, HID);
}
